// RelGraphConvHetero_61478161874989
// MI455X (gfx1250) — compile-verified, matrix operand order checked by reading, not run on hardware
//
#include <hip/hip_runtime.h>
#include <hip/hip_bf16.h>

// ---------------------------------------------------------------------------
// RGCN hetero layer for MI455X (gfx1250, wave32, WMMA)
//
// N0 = N1 = 50000, E = 160000 per relation, IN=OUT=128, BASES=2, RELS=4
//
// Pipeline:
//  1) init_bias      : d_out[..] = h_bias[col]
//  2) compose_pack   : W[r] = wc[r,0]*B0 + wc[r,1]*B1, packed f16 in the
//                      per-lane v_wmma_f32_16x16x32_f16 B-fragment layout
//  3) per relation r : gemm (X_src @ W[r] -> H, WMMA)  then scatter-add
// ---------------------------------------------------------------------------

typedef __attribute__((ext_vector_type(16))) _Float16 v16h;
typedef __attribute__((ext_vector_type(8)))  float    v8f;

#define NN        50000
#define EDGES     160000
#define FEAT      128
#define MTILES    3125            // 50000 / 16
#define WAVES_PB  8
#define FRAG_H    16              // halves per lane per fragment
#define FRAG_LANES 32
// one (kb,nb) B fragment = 32 lanes * 16 halves; per relation: 4 kb * 8 nb
#define WFRAG_PER_REL (4 * 8 * FRAG_LANES * FRAG_H)   // 16384 halves = 32 KB

// K index held by (vgpr-slot v, packed element e, lane-half h) for 16-bit
// 16x16x32 A/B fragments (ISA 7.12.2):
__device__ __forceinline__ int wmma_k_of(int v, int e, int h) {
    return (v < 4) ? (2 * v + e + 8 * h) : (16 + 2 * (v - 4) + e + 8 * h);
}

// ---------------- 1) bias broadcast into output ----------------------------
__global__ void rgcn_init_bias(float* __restrict__ out,
                               const float* __restrict__ bias,
                               long total) {
    long i = (long)blockIdx.x * blockDim.x + threadIdx.x;
    if (i < total) out[i] = bias[i & (FEAT - 1)];
}

// ---------------- 2) basis combine + pack into B-fragment layout -----------
// thread t -> (r, kb, nb, lane); writes 16 halves (one lane's fragment)
__global__ void rgcn_compose_pack(const float* __restrict__ weight,  // [2,128,128]
                                  const float* __restrict__ w_comp,  // [4,2]
                                  _Float16* __restrict__ wfrag) {
    int t = blockIdx.x * blockDim.x + threadIdx.x;
    if (t >= 4 * 4 * 8 * FRAG_LANES) return;
    int lane = t & 31;
    int nb   = (t >> 5) & 7;
    int kb   = (t >> 8) & 3;
    int r    = (t >> 10) & 3;
    int n = lane & 15;
    int h = lane >> 4;
    float c0 = w_comp[r * 2 + 0];
    float c1 = w_comp[r * 2 + 1];
    _Float16* o = wfrag + ((long)((r * 4 + kb) * 8 + nb) * FRAG_LANES + lane) * FRAG_H;
#pragma unroll
    for (int idx = 0; idx < FRAG_H; ++idx) {
        int v = idx >> 1, e = idx & 1;
        int K = kb * 32 + wmma_k_of(v, e, h);
        int N = nb * 16 + n;
        float val = c0 * weight[(long)K * FEAT + N]
                  + c1 * weight[(long)(FEAT * FEAT) + (long)K * FEAT + N];
        o[idx] = (_Float16)val;
    }
}

// ---------------- 3a) GEMM  H = X @ W_r  via v_wmma_f32_16x16x32_f16 -------
// one wave per 16-row M tile; A frags (4 K-blocks) kept in VGPRs,
// B frags streamed from the pre-packed (L2-resident) weight buffer.
__global__ void __launch_bounds__(WAVES_PB * 32)
rgcn_gemm_wmma(const float* __restrict__ X,        // [50000,128] f32
               const _Float16* __restrict__ wfr,   // one relation's fragments
               float* __restrict__ H) {            // [50000,128] f32
    int wave = threadIdx.x >> 5;
    int lane = threadIdx.x & 31;
    int mt = blockIdx.x * WAVES_PB + wave;
    if (mt >= MTILES) return;                       // wave-uniform exit

    int m = lane & 15;
    int h = lane >> 4;
    const float* xrow = X + ((long)mt * 16 + m) * FEAT;

    // A fragments: lane (m,h) holds K = kb*32 + {8h..8h+7} U {16+8h..16+8h+7}
    v16h a[4];
#pragma unroll
    for (int kb = 0; kb < 4; ++kb) {
        const float* p0 = xrow + kb * 32 + 8 * h;
        const float* p1 = p0 + 16;
#pragma unroll
        for (int j = 0; j < 8; ++j) {
            a[kb][j]     = (_Float16)p0[j];
            a[kb][8 + j] = (_Float16)p1[j];
        }
    }

    const v16h* bf = reinterpret_cast<const v16h*>(wfr);
    float* hout = H + (long)mt * 16 * FEAT;

#pragma unroll
    for (int nb = 0; nb < 8; ++nb) {
        v8f c = {};
#pragma unroll
        for (int kb = 0; kb < 4; ++kb) {
            v16h b = bf[(kb * 8 + nb) * FRAG_LANES + lane];
            c = __builtin_amdgcn_wmma_f32_16x16x32_f16(
                    /*neg_a=*/false, a[kb], /*neg_b=*/false, b,
                    /*c_mod=*/(short)0, c, /*reuse_a=*/false, /*reuse_b=*/false);
        }
        // C layout: VGPR v -> row (8*h + v), col = nb*16 + (lane&15)
#pragma unroll
        for (int v = 0; v < 8; ++v) {
            hout[(long)(8 * h + v) * FEAT + nb * 16 + m] = c[v];
        }
    }
}

// ---------------- 3b) edge gather + scatter-add ----------------------------
// 128 threads per edge; f32 atomic add (non-returning -> global_atomic_add_f32)
__global__ void rgcn_scatter(const float* __restrict__ H,
                             const int* __restrict__ src,
                             const int* __restrict__ dst,
                             float* __restrict__ out) {
    long t = (long)blockIdx.x * blockDim.x + threadIdx.x;
    long e = t >> 7;
    int  c = (int)(t & 127);
    if (e < EDGES) {
        int s = src[e];
        int d = dst[e];
        float v = H[(long)s * FEAT + c];
        atomicAdd(out + (long)d * FEAT + c, v);
    }
}

// ---------------------------------------------------------------------------
extern "C" void kernel_launch(void* const* d_in, const int* in_sizes, int n_in,
                              void* d_out, int out_size, void* d_ws, size_t ws_size,
                              hipStream_t stream) {
    const float* x0     = (const float*)d_in[0];
    const float* x1     = (const float*)d_in[1];
    const float* weight = (const float*)d_in[2];
    const float* w_comp = (const float*)d_in[3];
    const float* h_bias = (const float*)d_in[4];
    const int* src0 = (const int*)d_in[5];
    const int* dst0 = (const int*)d_in[6];
    const int* src1 = (const int*)d_in[7];
    const int* dst1 = (const int*)d_in[8];
    const int* src2 = (const int*)d_in[9];
    const int* dst2 = (const int*)d_in[10];
    const int* src3 = (const int*)d_in[11];
    const int* dst3 = (const int*)d_in[12];

    float* out0 = (float*)d_out;                        // h0: [50000,128]
    float* out1 = out0 + (long)NN * FEAT;               // h1: [50000,128]

    char* ws = (char*)d_ws;
    _Float16* wfrag = (_Float16*)ws;                    // 4 * 32 KB packed weights
    float* H = (float*)(ws + 4 * WFRAG_PER_REL * sizeof(_Float16)); // 25.6 MB

    // 1) bias into both output halves
    {
        long total = 2L * NN * FEAT;
        int blocks = (int)((total + 255) / 256);
        rgcn_init_bias<<<blocks, 256, 0, stream>>>(out0, h_bias, total);
    }
    // 2) compose + pack all 4 relation weight matrices
    rgcn_compose_pack<<<16, 256, 0, stream>>>(weight, w_comp, wfrag);

    const int gemm_blocks = (MTILES + WAVES_PB - 1) / WAVES_PB;       // 391
    const int scat_blocks = (int)(((long)EDGES * FEAT + 255) / 256);  // 80000

    // relation 0: x0 -> n0
    rgcn_gemm_wmma<<<gemm_blocks, WAVES_PB * 32, 0, stream>>>(x0, wfrag + 0 * WFRAG_PER_REL, H);
    rgcn_scatter<<<scat_blocks, 256, 0, stream>>>(H, src0, dst0, out0);
    // relation 1: x0 -> n1
    rgcn_gemm_wmma<<<gemm_blocks, WAVES_PB * 32, 0, stream>>>(x0, wfrag + 1 * WFRAG_PER_REL, H);
    rgcn_scatter<<<scat_blocks, 256, 0, stream>>>(H, src1, dst1, out1);
    // relation 2: x1 -> n0
    rgcn_gemm_wmma<<<gemm_blocks, WAVES_PB * 32, 0, stream>>>(x1, wfrag + 2 * WFRAG_PER_REL, H);
    rgcn_scatter<<<scat_blocks, 256, 0, stream>>>(H, src2, dst2, out0);
    // relation 3: x1 -> n1
    rgcn_gemm_wmma<<<gemm_blocks, WAVES_PB * 32, 0, stream>>>(x1, wfrag + 3 * WFRAG_PER_REL, H);
    rgcn_scatter<<<scat_blocks, 256, 0, stream>>>(H, src3, dst3, out1);
}
